// MultiHeadAttentionSpatial_65335042506984
// MI455X (gfx1250) — matrix operand order repaired and checked by reading, not compiled, not run on hardware
//
#include <hip/hip_runtime.h>
#include <hip/hip_bf16.h>
#include <math.h>

// ---------------- problem constants ----------------
#define B_  8
#define L_  512
#define T_  512
#define D_  768
#define H_  12
#define SD_ 5
#define DH_ 64          // D / H
#define ML_ (B_ * L_)   // 4096 rows for projections

typedef __attribute__((ext_vector_type(16))) __bf16 v16bf;
typedef __attribute__((ext_vector_type(8)))  float  v8f;

union FragBF {
    v16bf v;
    unsigned int u[8];
    unsigned short s[16];
};

static __device__ __forceinline__ unsigned short f2bf(float f) {
    unsigned int u = __float_as_uint(f);
    unsigned int r = u + 0x7FFFu + ((u >> 16) & 1u);   // round-to-nearest-even
    return (unsigned short)(r >> 16);
}

// A-matrix 16x32 bf16 fragment (ISA 7.12.2): lanes 0-15 = rows M0-15 hold
// K=0..7 (VGPR0-3) and K=16..23 (VGPR4-7); lanes 16-31 hold K=8..15 / 24..31.
static __device__ __forceinline__ v16bf load_frag_a(const unsigned short* base, int ldr) {
    const int lane = threadIdx.x & 31;
    const int r    = lane & 15;
    const int kh   = (lane >> 4) * 8;
    FragBF f;
    const unsigned short* p = base + r * ldr;
#pragma unroll
    for (int j = 0; j < 4; ++j)
        f.u[j]     = *(const unsigned int*)(p + kh + 2 * j);
#pragma unroll
    for (int j = 0; j < 4; ++j)
        f.u[4 + j] = *(const unsigned int*)(p + 16 + kh + 2 * j);
    return f.v;
}

// B-matrix 32x16 bf16 fragment: base points at B^T (row n = column n of B,
// contiguous in k). Lanes 0-15 hold K=0..7 / 16..23, lanes 16-31 hold K=8..15 / 24..31.
static __device__ __forceinline__ v16bf load_frag_b(const unsigned short* base, int ldr) {
    const int lane = threadIdx.x & 31;
    const int r    = lane & 15;
    const int kb   = (lane >> 4) * 8;
    FragBF f;
    const unsigned short* p = base + r * ldr + kb;
#pragma unroll
    for (int j = 0; j < 8; ++j)
        f.u[j] = *(const unsigned int*)(p + 2 * j + (j >> 2) * 8);
    return f.v;
}

static __device__ __forceinline__ v8f wmma_bf16(v16bf a, v16bf b, v8f c) {
    return __builtin_amdgcn_wmma_f32_16x16x32_bf16(false, a, false, b, (short)0, c, false, false);
}

// ---------------- Tensor Data Mover (TDM) support ----------------
#if defined(__has_builtin)
#if __has_builtin(__builtin_amdgcn_tensor_load_to_lds) && \
    __has_builtin(__builtin_amdgcn_s_wait_tensorcnt)
#define USE_TDM 1
#endif
#endif

#ifdef USE_TDM
typedef __attribute__((ext_vector_type(4))) unsigned int u32x4;
typedef __attribute__((ext_vector_type(8))) int i32x8;
typedef __attribute__((ext_vector_type(4))) int i32x4;

typedef __attribute__((address_space(3))) void lds_void_t;

static __device__ __forceinline__ unsigned lds_byte_off(const void* p) {
    return (unsigned)(unsigned long long)(lds_void_t*)p;
}

// Issue a 2D f32 tile DMA: global (rows x row_elems, row stride in elems)
// -> LDS packed row-major. Per ISA 8.3/8.4 D# layout.
static __device__ __forceinline__ void tdm_load_2d_f32(const float* gsrc,
                                                       unsigned lds_off,
                                                       int row_elems, int rows,
                                                       long long row_stride_elems) {
    const unsigned long long ga = (unsigned long long)gsrc;
    // group0: [1:0]=count=1, [63:32]=lds_addr, [120:64]=global_addr, [127:126]=type=2
    u32x4 g0 = { 1u,
                 lds_off,
                 (unsigned)(ga & 0xFFFFFFFFu),
                 (unsigned)((ga >> 32) & 0x01FFFFFFu) | (2u << 30) };
    const unsigned long long s0 = (unsigned long long)row_stride_elems;
    const unsigned td0 = 1u << 30, td1 = 1u << 30;      // huge -> no OOB clamp
    i32x8 g1 = { (int)(2u << 16),                                   // data_size=4B
                 (int)((td0 & 0xFFFFu) << 16),                      // tensor_dim0 lo
                 (int)((td0 >> 16) | ((td1 & 0xFFFFu) << 16)),      // dim0 hi | dim1 lo
                 (int)((td1 >> 16) | ((unsigned)row_elems << 16)),  // dim1 hi | tile_dim0
                 (int)((unsigned)rows),                             // tile_dim1 | tile_dim2=0
                 (int)(s0 & 0xFFFFFFFFull),                         // dim0_stride lo
                 (int)((s0 >> 32) & 0xFFFFull),                     // dim0_stride hi | dim1_stride lo
                 0 };                                               // dim1_stride hi
    i32x4 gz = { 0, 0, 0, 0 };
#if __clang_major__ >= 23
    i32x8 gz8 = { 0, 0, 0, 0, 0, 0, 0, 0 };
    __builtin_amdgcn_tensor_load_to_lds(g0, g1, gz, gz, gz8, 0);
#else
    __builtin_amdgcn_tensor_load_to_lds(g0, g1, gz, gz, 0);
#endif
}
#endif // USE_TDM

// ---------------------------------------------------------------------------
// Kernel 1/6: tiled GEMM  Out = X(MxK,f32) @ W(KxN,f32) + bias, WMMA bf16.
// Block = 256 threads = 8 waves; block tile 32(M) x 64(N); K-step 32.
// ---------------------------------------------------------------------------
template <bool OUT_BF16>
__global__ void gemm_rowmajor(const float* __restrict__ X,
                              const float* __restrict__ W,
                              const float* __restrict__ bias,
                              void* __restrict__ Out,
                              int M, int N, int K) {
    __shared__ unsigned short As[32 * 34];   // [m][k], stride 34
    __shared__ unsigned short Bs[64 * 34];   // [n][k] = W^T tile, stride 34

    const int tid  = threadIdx.x;
    const int m0   = blockIdx.y * 32;
    const int n0   = blockIdx.x * 64;
    const int wave = tid >> 5;
    const int wm   = wave & 1;               // 0..1 -> 16-row slice
    const int wn   = wave >> 1;              // 0..3 -> 16-col slice

    v8f acc = {0.f, 0.f, 0.f, 0.f, 0.f, 0.f, 0.f, 0.f};

    for (int k0 = 0; k0 < K; k0 += 32) {
        // stage A tile 32x32 (f32 -> bf16)
        {
            const int row  = tid >> 3;        // 0..31
            const int kcol = (tid & 7) * 4;   // 0..28
            const float* xp = X + (size_t)(m0 + row) * K + k0 + kcol;
            if (k0 + 32 < K) __builtin_prefetch(xp + 32, 0, 1);  // global_prefetch_b8
#pragma unroll
            for (int i = 0; i < 4; ++i)
                As[row * 34 + kcol + i] = f2bf(xp[i]);
        }
        // stage B tile transposed: Bs[n][k] = W[k0+k][n0+n]
        {
            const int n    = tid >> 2;        // 0..63
            const int kcol = (tid & 3) * 8;   // 0..24
#pragma unroll
            for (int i = 0; i < 8; ++i)
                Bs[n * 34 + kcol + i] = f2bf(W[(size_t)(k0 + kcol + i) * N + n0 + n]);
        }
        __syncthreads();

        v16bf a = load_frag_a(As + wm * 16 * 34, 34);
        v16bf b = load_frag_b(Bs + wn * 16 * 34, 34);
        acc = wmma_bf16(a, b, acc);
        __syncthreads();
    }

    // C layout: lane<16 -> M = j, lane>=16 -> M = j+8; N = lane & 15.
    const int lane = tid & 31;
    const int colL = lane & 15;
    const int rowb = (lane >> 4) * 8;
    const int gn   = n0 + wn * 16 + colL;
    const float bv = bias[gn];
#pragma unroll
    for (int j = 0; j < 8; ++j) {
        const int gm = m0 + wm * 16 + rowb + j;
        const float val = acc[j] + bv;
        if (OUT_BF16)
            ((unsigned short*)Out)[(size_t)gm * N + gn] = f2bf(val);
        else
            ((float*)Out)[(size_t)gm * N + gn] = val;
    }
}

// ---------------------------------------------------------------------------
// Kernel 2: SW = q @ Wl + bl   (N = 72, too thin for WMMA tiles)
// ---------------------------------------------------------------------------
__global__ void sw_proj(const float* __restrict__ q,
                        const float* __restrict__ Wl,
                        const float* __restrict__ bl,
                        float* __restrict__ SW) {
    const int idx = blockIdx.x * blockDim.x + threadIdx.x;
    const int NW  = H_ * (SD_ + 1);          // 72
    if (idx >= ML_ * NW) return;
    const int m = idx / NW;
    const int n = idx - m * NW;
    const float* xq = q + (size_t)m * D_;
    float acc = bl[n];
#pragma unroll 4
    for (int k = 0; k < D_; ++k)
        acc = fmaf(xq[k], Wl[(size_t)k * NW + n], acc);
    SW[idx] = acc;
}

// ---------------------------------------------------------------------------
// Kernel 3: fused logits  z = log(clip(sigmoid(spatial))) + QK^T/8, masked.
// Grid (T/16, L/16, B); block 128 = 4 waves, each wave serves 3 heads.
// pairwise_locs tile and SW tile staged once per block via the Tensor Data
// Mover (TDM) when available, reused by all 12 heads.
// log(clip(sigmoid(s),1e-6)) = max(-log(1+exp(-s)), log(1e-6)) -> no fp divide.
// ---------------------------------------------------------------------------
__global__ void logits_kernel(const unsigned short* __restrict__ Qbf,
                              const unsigned short* __restrict__ Kbf,
                              const float* __restrict__ SW,
                              const float* __restrict__ plocsG,
                              const unsigned char* __restrict__ maskG,
                              float* __restrict__ fused) {
    __shared__ float plocs[16 * 80];         // [m][n*5+s]
    __shared__ float sws[16 * 72];           // [m][h*6+c]
    __shared__ unsigned char mk[16];

    const int tid = threadIdx.x;
    const int t0  = blockIdx.x * 16;
    const int l0  = blockIdx.y * 16;
    const int b   = blockIdx.z;

#ifdef USE_TDM
    if (tid < 32) {                           // wave 0 drives the TDM
        // pairwise_locs tile: 16 rows of 80 contiguous f32, row stride T*SD
        tdm_load_2d_f32(plocsG + (((size_t)(b * L_ + l0)) * T_ + t0) * SD_,
                        lds_byte_off(plocs), 80, 16, (long long)T_ * SD_);
        // SW tile: 16 rows of 72 contiguous f32, row stride 72
        tdm_load_2d_f32(SW + (size_t)(b * L_ + l0) * 72,
                        lds_byte_off(sws), 72, 16, 72);
        if (tid < 16) mk[tid] = maskG[b * T_ + t0 + tid];
        __builtin_amdgcn_s_wait_tensorcnt(0);
    }
#else
#pragma unroll
    for (int i = 0; i < 10; ++i) {           // 1280 floats
        const int f = tid + i * 128;
        const int m = f / 80, rem = f - m * 80;
        plocs[f] = plocsG[(((size_t)(b * L_ + l0 + m)) * T_ + t0 + rem / 5) * SD_ + (rem % 5)];
    }
#pragma unroll
    for (int i = 0; i < 9; ++i) {            // 1152 floats
        const int f = tid + i * 128;
        const int m = f / 72, c = f - m * 72;
        sws[f] = SW[(size_t)(b * L_ + l0 + m) * 72 + c];
    }
    if (tid < 16) mk[tid] = maskG[b * T_ + t0 + tid];
#endif
    __syncthreads();

    const int wave = tid >> 5;
    const int lane = tid & 31;
    const int colL = lane & 15;
    const int rowb = (lane >> 4) * 8;
    const bool msk = mk[colL] != 0;
    const float LOG_CLIP = -13.815511f;      // log(1e-6)

    for (int hi = 0; hi < 3; ++hi) {
        const int h = wave + 4 * hi;         // 0..11
        const unsigned short* qb = Qbf + (size_t)(b * L_ + l0) * D_ + h * DH_;
        const unsigned short* kb = Kbf + (size_t)(b * T_ + t0) * D_ + h * DH_;

        v8f acc = {0.f, 0.f, 0.f, 0.f, 0.f, 0.f, 0.f, 0.f};
#pragma unroll
        for (int ks = 0; ks < 2; ++ks) {     // dh=64 -> two K=32 steps
            v16bf a = load_frag_a(qb + ks * 32, D_);
            v16bf bfr = load_frag_b(kb + ks * 32, D_);
            acc = wmma_bf16(a, bfr, acc);
        }

#pragma unroll
        for (int j = 0; j < 8; ++j) {
            const int m = rowb + j;
            float z;
            if (msk) {
                z = -__builtin_inff();
            } else {
                const float attn = acc[j] * 0.125f;     // 1/sqrt(64)
                const float* wv = &sws[m * 72 + h * 6]; // [bias, w0..w4]
                const float* pl = &plocs[m * 80 + colL * 5];
                float s = wv[0];
#pragma unroll
                for (int si = 0; si < SD_; ++si) s = fmaf(wv[1 + si], pl[si], s);
                // log(sigmoid(s)) = -log(1 + exp(-s)); clip below at log(1e-6)
                const float logsig = -logf(1.0f + expf(-s));
                z = fmaxf(logsig, LOG_CLIP) + attn;
            }
            fused[(((size_t)(h * B_ + b) * L_ + l0 + m)) * T_ + t0 + colL] = z;
        }
    }
}

// ---------------------------------------------------------------------------
// Kernel 4: row softmax over T=512. One wave per row, 8 rows per block.
// ---------------------------------------------------------------------------
__global__ void softmax_rows(float* __restrict__ fused) {
    const int wave = threadIdx.x >> 5;
    const int lane = threadIdx.x & 31;
    const int row  = blockIdx.x * 8 + wave;
    float* p = fused + (size_t)row * T_;

    float v[16];
    float m = -__builtin_inff();
#pragma unroll
    for (int i = 0; i < 16; ++i) { v[i] = p[lane + i * 32]; m = fmaxf(m, v[i]); }
#pragma unroll
    for (int off = 16; off > 0; off >>= 1) m = fmaxf(m, __shfl_xor(m, off, 32));

    float s = 0.f;
    const bool dead = (m == -__builtin_inff());
#pragma unroll
    for (int i = 0; i < 16; ++i) {
        v[i] = dead ? 0.f : expf(v[i] - m);
        s += v[i];
    }
#pragma unroll
    for (int off = 16; off > 0; off >>= 1) s += __shfl_xor(s, off, 32);
    const float inv = (s > 0.f) ? 1.0f / s : 0.f;
#pragma unroll
    for (int i = 0; i < 16; ++i) p[lane + i * 32] = v[i] * inv;
}

// ---------------------------------------------------------------------------
// Kernel 5: out_heads[l, h*64+dv] = sum_t fused[h,b,l,t] * Vh[t,dv]  (WMMA)
// Grid (L/16, B, H); block 128 = 4 waves over the 64 dv columns.
// ---------------------------------------------------------------------------
__global__ void attn_v_kernel(const float* __restrict__ fused,
                              const unsigned short* __restrict__ Vbf,
                              float* __restrict__ OutHeads) {
    __shared__ unsigned short Vt[64 * 34];   // [dv][t] transposed tile, K-step 32

    const int tid = threadIdx.x;
    const int l0  = blockIdx.x * 16;
    const int b   = blockIdx.y;
    const int h   = blockIdx.z;
    const int wn  = tid >> 5;
    const int lane = tid & 31;
    const int r    = lane & 15;
    const int kh   = (lane >> 4) * 8;

    v8f acc = {0.f, 0.f, 0.f, 0.f, 0.f, 0.f, 0.f, 0.f};

    for (int t0 = 0; t0 < T_; t0 += 32) {
        // stage V tile transposed: Vt[dv][k] = Vbf[b*T + t0+k][h*64+dv]
        {
            const int k   = tid >> 2;         // 0..31
            const int dvb = (tid & 3) * 16;   // 0..48
            const unsigned short* vp = Vbf + (size_t)(b * T_ + t0 + k) * D_ + h * DH_ + dvb;
#pragma unroll
            for (int i = 0; i < 16; ++i)
                Vt[(dvb + i) * 34 + k] = vp[i];
        }
        __syncthreads();

        // A fragment from fp32 probabilities (convert in-register)
        FragBF fa;
        const float* fp = fused + (((size_t)(h * B_ + b) * L_ + l0 + r)) * T_ + t0;
#pragma unroll
        for (int j = 0; j < 4; ++j) {
            fa.s[2 * j]     = f2bf(fp[kh + 2 * j]);
            fa.s[2 * j + 1] = f2bf(fp[kh + 2 * j + 1]);
        }
#pragma unroll
        for (int j = 0; j < 4; ++j) {
            fa.s[8 + 2 * j]     = f2bf(fp[16 + kh + 2 * j]);
            fa.s[8 + 2 * j + 1] = f2bf(fp[16 + kh + 2 * j + 1]);
        }
        v16bf bfr = load_frag_b(Vt + wn * 16 * 34, 34);
        acc = wmma_bf16(fa.v, bfr, acc);
        __syncthreads();
    }

    const int colL = lane & 15;
    const int rowb = (lane >> 4) * 8;
#pragma unroll
    for (int j = 0; j < 8; ++j)
        OutHeads[(size_t)(b * L_ + l0 + rowb + j) * D_ + h * DH_ + wn * 16 + colL] = acc[j];
}

// ---------------------------------------------------------------------------
extern "C" void kernel_launch(void* const* d_in, const int* in_sizes, int n_in,
                              void* d_out, int out_size, void* d_ws, size_t ws_size,
                              hipStream_t stream) {
    const float* q   = (const float*)d_in[0];
    const float* k   = (const float*)d_in[1];
    const float* v   = (const float*)d_in[2];
    const float* pl  = (const float*)d_in[3];
    const unsigned char* mask = (const unsigned char*)d_in[4];
    const float* Wq  = (const float*)d_in[5];
    const float* bq  = (const float*)d_in[6];
    const float* Wk  = (const float*)d_in[7];
    const float* bk  = (const float*)d_in[8];
    const float* Wv  = (const float*)d_in[9];
    const float* bv  = (const float*)d_in[10];
    const float* Wl  = (const float*)d_in[11];
    const float* bl  = (const float*)d_in[12];
    const float* Wf  = (const float*)d_in[13];
    const float* bf  = (const float*)d_in[14];

    float* out   = (float*)d_out;                               // (B,L,D)
    float* fused = (float*)d_out + (size_t)ML_ * D_;            // (H,B,L,T)

    char* ws = (char*)d_ws;
    unsigned short* Qbf = (unsigned short*)ws;                  // 6 MB
    unsigned short* Kbf = (unsigned short*)(ws + 6291456);      // 6 MB
    unsigned short* Vbf = (unsigned short*)(ws + 12582912);     // 6 MB
    float* SW           = (float*)(ws + 18874368);              // 1.2 MB
    float* OutHeads     = (float*)(ws + 20054016);              // 12.6 MB

    const dim3 gGemm(D_ / 64, ML_ / 32);
    gemm_rowmajor<true><<<gGemm, 256, 0, stream>>>(q, Wq, bq, Qbf, ML_, D_, D_);
    gemm_rowmajor<true><<<gGemm, 256, 0, stream>>>(k, Wk, bk, Kbf, ML_, D_, D_);
    gemm_rowmajor<true><<<gGemm, 256, 0, stream>>>(v, Wv, bv, Vbf, ML_, D_, D_);

    sw_proj<<<(ML_ * 72 + 255) / 256, 256, 0, stream>>>(q, Wl, bl, SW);

    logits_kernel<<<dim3(T_ / 16, L_ / 16, B_), 128, 0, stream>>>(
        Qbf, Kbf, SW, pl, mask, fused);

    softmax_rows<<<(H_ * B_ * L_) / 8, 256, 0, stream>>>(fused);

    attn_v_kernel<<<dim3(L_ / 16, B_, H_), 128, 0, stream>>>(fused, Vbf, OutHeads);

    gemm_rowmajor<false><<<gGemm, 256, 0, stream>>>(OutHeads, Wf, bf, out, ML_, D_, D_);
}
